// DualSTAGE_74955769249861
// MI455X (gfx1250) — compile-verified, matrix operand order checked by reading, not run on hardware
//
#include <hip/hip_runtime.h>
#include <math.h>

typedef __attribute__((ext_vector_type(16))) _Float16 v16h;
typedef __attribute__((ext_vector_type(8)))  float    v8f;

#define Bb   2
#define Nn   1024
#define Cc   128
#define Ee   64
#define Kt   20
#define ROWS (Bb * Nn)
#define BNK  (ROWS * Kt)

// ---------------------------------------------------------------------------
// Kernel 1: x_l = x@Wl + bl ; x_r = x@Wr + br  via v_wmma_f32_16x16x32_f16.
// One wave handles one 16-row M tile and all 4 E-column tiles (E=64).
// blockIdx.y selects left/right weight matrix.
// ---------------------------------------------------------------------------
__global__ __launch_bounds__(128) void gat_gemm_wmma(
    const float* __restrict__ x,
    const float* __restrict__ Wl, const float* __restrict__ bl,
    const float* __restrict__ Wr, const float* __restrict__ br,
    float* __restrict__ xl, float* __restrict__ xr)
{
    const int lane = threadIdx.x & 31;
    const int wave = threadIdx.x >> 5;
    const int mblk = blockIdx.x * 4 + wave;      // 0..127
    const int m0   = mblk * 16;
    if (m0 >= ROWS) return;

    const bool right = (blockIdx.y != 0);
    const float* __restrict__ W    = right ? Wr : Wl;
    const float* __restrict__ bias = right ? br : bl;
    float* __restrict__ out        = right ? xr : xl;

    const int half = lane >> 4;      // which K-half this lane covers
    const int lcol = lane & 15;      // M (for A) / N (for B,C,D) index

    const float* __restrict__ arow = x + (size_t)(m0 + lcol) * Cc;

    v8f acc[4];
#pragma unroll
    for (int et = 0; et < 4; ++et)
        acc[et] = (v8f){0.f, 0.f, 0.f, 0.f, 0.f, 0.f, 0.f, 0.f};

#pragma unroll
    for (int kk = 0; kk < 4; ++kk) {             // K = 128 in steps of 32
        const int kb = kk * 32;
        // A fragment (16-bit A 16x32 layout): lane holds row m0+lcol,
        //   a[0..7]  = K = kb + 8*half + (0..7)
        //   a[8..15] = K = kb + 16 + 8*half + (0..7)
        const float4 p0 = *(const float4*)(arow + kb + 8 * half);
        const float4 p1 = *(const float4*)(arow + kb + 8 * half + 4);
        const float4 p2 = *(const float4*)(arow + kb + 16 + 8 * half);
        const float4 p3 = *(const float4*)(arow + kb + 16 + 8 * half + 4);
        v16h a;
        a[0]  = (_Float16)p0.x; a[1]  = (_Float16)p0.y;
        a[2]  = (_Float16)p0.z; a[3]  = (_Float16)p0.w;
        a[4]  = (_Float16)p1.x; a[5]  = (_Float16)p1.y;
        a[6]  = (_Float16)p1.z; a[7]  = (_Float16)p1.w;
        a[8]  = (_Float16)p2.x; a[9]  = (_Float16)p2.y;
        a[10] = (_Float16)p2.z; a[11] = (_Float16)p2.w;
        a[12] = (_Float16)p3.x; a[13] = (_Float16)p3.y;
        a[14] = (_Float16)p3.z; a[15] = (_Float16)p3.w;

#pragma unroll
        for (int et = 0; et < 4; ++et) {         // 4 column tiles of 16
            const int col = et * 16 + lcol;
            v16h bm;
#pragma unroll
            for (int rr = 0; rr < 8; ++rr) {
                const int k0 = kb + ((rr < 4) ? (8 * half + 2 * rr)
                                              : (16 + 8 * half + 2 * (rr - 4)));
                bm[2 * rr]     = (_Float16)W[(size_t)k0 * Ee + col];
                bm[2 * rr + 1] = (_Float16)W[(size_t)(k0 + 1) * Ee + col];
            }
            acc[et] = __builtin_amdgcn_wmma_f32_16x16x32_f16(
                false, a, false, bm, (short)0, acc[et], false, false);
        }
    }

    // D layout: VGPR r -> M = r + 8*half, N = lcol
#pragma unroll
    for (int et = 0; et < 4; ++et) {
        const int col = et * 16 + lcol;
        const float bv = bias[col];
#pragma unroll
        for (int rr = 0; rr < 8; ++rr) {
            const int row = m0 + rr + 8 * half;
            out[(size_t)row * Ee + col] = acc[et][rr] + bv;
        }
    }
}

// ---------------------------------------------------------------------------
// Kernel 2: fused pairwise score + nan_to_num + top-20 + softmax + indices.
// One wave per output row i (8 rows per 256-thread workgroup).
// 1024 scores per row stay in LDS (never touch HBM).
// ---------------------------------------------------------------------------
__global__ __launch_bounds__(256) void gat_topk_softmax(
    const float* __restrict__ xl, const float* __restrict__ xr,
    const float* __restrict__ att, float* __restrict__ out)
{
    __shared__ float s_alpha[8][Nn];   // 32 KB
    __shared__ float s_xl[8][Ee];
    __shared__ float s_att[Ee];
    __shared__ float s_topv[8][Kt];
    __shared__ int   s_topi[8][Kt];

    const int lane = threadIdx.x & 31;
    const int w    = threadIdx.x >> 5;
    const int r    = blockIdx.x * 8 + w;     // global row  (= b*N + i)
    const int b    = r >> 10;

    if (threadIdx.x < Ee) s_att[threadIdx.x] = att[threadIdx.x];
    s_xl[w][lane]      = xl[(size_t)r * Ee + lane];
    s_xl[w][lane + 32] = xl[(size_t)r * Ee + lane + 32];
    __syncthreads();

    const float* __restrict__ xrb = xr + (size_t)b * Nn * Ee;

    // Phase A: alpha[j] = sum_d att[d] * lrelu(xl[d] + xr[j][d]),  lrelu=max(s,0.2s)
    for (int t = 0; t < Nn / 32; ++t) {
        const int j = t * 32 + lane;
        const float4* __restrict__ p = (const float4*)(xrb + (size_t)j * Ee);
        float acc = 0.f;
#pragma unroll
        for (int d4 = 0; d4 < Ee / 4; ++d4) {
            const float4 q = p[d4];
            float s;
            s = s_xl[w][4 * d4 + 0] + q.x; acc += s_att[4 * d4 + 0] * fmaxf(s, 0.2f * s);
            s = s_xl[w][4 * d4 + 1] + q.y; acc += s_att[4 * d4 + 1] * fmaxf(s, 0.2f * s);
            s = s_xl[w][4 * d4 + 2] + q.z; acc += s_att[4 * d4 + 2] * fmaxf(s, 0.2f * s);
            s = s_xl[w][4 * d4 + 3] + q.w; acc += s_att[4 * d4 + 3] * fmaxf(s, 0.2f * s);
        }
        if (!__builtin_isfinite(acc)) acc = 0.f;   // nan_to_num
        s_alpha[w][j] = acc;
    }

    // Phase B: iterative exact top-K (stable: lowest j wins ties, like lax.top_k)
#pragma unroll 1
    for (int kk = 0; kk < Kt; ++kk) {
        float best = -INFINITY;
        int   bj   = 0x7fffffff;
        for (int t = 0; t < Nn / 32; ++t) {
            const int j = lane + t * 32;           // ascending j per lane
            const float v = s_alpha[w][j];
            if (v > best) { best = v; bj = j; }    // keeps first max per lane
        }
#pragma unroll
        for (int off = 16; off > 0; off >>= 1) {
            const float ov = __shfl_xor(best, off, 32);
            const int   oj = __shfl_xor(bj, off, 32);
            if (ov > best || (ov == best && oj < bj)) { best = ov; bj = oj; }
        }
        if (lane == 0) {
            s_topv[w][kk] = best;
            s_topi[w][kk] = bj;
            s_alpha[w][bj] = -INFINITY;            // remove winner
        }
        // same-wave LDS ops are in-order; compiler inserts s_wait_dscnt
    }

    // Phase C: softmax over the 20 (descending) values + emit indices
    const float v  = (lane < Kt) ? s_topv[w][lane] : -INFINITY;
    float mx = v;
#pragma unroll
    for (int off = 16; off > 0; off >>= 1)
        mx = fmaxf(mx, __shfl_xor(mx, off, 32));
    float ex = (lane < Kt) ? __expf(v - mx) : 0.f;
    float sum = ex;
#pragma unroll
    for (int off = 16; off > 0; off >>= 1)
        sum += __shfl_xor(sum, off, 32);

    if (lane < Kt) {
        const size_t base = (size_t)r * Kt + lane;
        out[base]           = (float)r;                              // index_i
        out[BNK + base]     = (float)(s_topi[w][lane] + b * Nn);     // index_j
        out[2 * BNK + base] = ex / sum;                              // attention
    }
}

extern "C" void kernel_launch(void* const* d_in, const int* in_sizes, int n_in,
                              void* d_out, int out_size, void* d_ws, size_t ws_size,
                              hipStream_t stream) {
    (void)in_sizes; (void)n_in; (void)out_size; (void)ws_size;
    const float* x   = (const float*)d_in[0];
    const float* Wl  = (const float*)d_in[1];
    const float* bl  = (const float*)d_in[2];
    const float* Wr  = (const float*)d_in[3];
    const float* br  = (const float*)d_in[4];
    const float* att = (const float*)d_in[5];
    // d_in[6] edge_index (int64), d_in[7] batch (int64): unused in eval path

    float* xl = (float*)d_ws;                       // [ROWS, 64]  512 KB
    float* xr = xl + (size_t)ROWS * Ee;             // [ROWS, 64]  512 KB

    dim3 g1(32, 2);                                 // 128 M-tiles x {l,r}
    gat_gemm_wmma<<<g1, 128, 0, stream>>>(x, Wl, bl, Wr, br, xl, xr);

    gat_topk_softmax<<<ROWS / 8, 256, 0, stream>>>(xl, xr, att, (float*)d_out);
}
